// WindowTransformerBlock_28930899706559
// MI455X (gfx1250) — compile-verified, matrix operand order checked by reading, not run on hardware
//
#include <hip/hip_runtime.h>
#include <hip/hip_bf16.h>

// ---------------------------------------------------------------------------
// Types for CDNA5 WMMA
// ---------------------------------------------------------------------------
typedef _Float16 h8   __attribute__((ext_vector_type(8)));
typedef _Float16 v16h __attribute__((ext_vector_type(16)));
typedef float    v8f  __attribute__((ext_vector_type(8)));

// Load a 16x32 f16 WMMA A-fragment (or 32x16 B-fragment from an N-major
// [n][k] buffer) from `base` with leading dimension `ld` (in halves).
// Per ISA layout: lanes 0-15 hold K {0..7,16..23}, lanes 16-31 hold
// K {8..15,24..31}; row = row0 + (lane&15).
__device__ __forceinline__ v16h ld_frag(const _Float16* base, int row0, int ld,
                                        int k0, int lane) {
  int r   = row0 + (lane & 15);
  int sel = (lane >> 4) << 3;                 // 0 or 8
  const _Float16* p = base + (size_t)r * ld + k0 + sel;
  h8 lo = *(const h8*)p;
  h8 hi = *(const h8*)(p + 16);
  return __builtin_shufflevector(lo, hi, 0, 1, 2, 3, 4, 5, 6, 7,
                                 8, 9, 10, 11, 12, 13, 14, 15);
}

__device__ __forceinline__ v8f wmma_f16(v16h a, v16h b, v8f c) {
  return __builtin_amdgcn_wmma_f32_16x16x32_f16(false, a, false, b,
                                                (short)0, c, false, false);
}

__device__ __forceinline__ int regi(int r) {
  return (r < 248) ? 0 : ((r < 252) ? 1 : 2);
}

// CDNA5 async global->LDS copy (16B per lane), tracked with ASYNCcnt.
// lds_off is the byte offset within the workgroup LDS allocation (dynamic
// segment starts at 0: no static __shared__ in these kernels).
__device__ __forceinline__ void async_g2l_b128(unsigned lds_off,
                                               const void* gptr) {
  asm volatile("global_load_async_to_lds_b128 %0, %1, off"
               :
               : "v"(lds_off), "v"((unsigned long long)(uintptr_t)gptr)
               : "memory");
}
__device__ __forceinline__ void wait_async0() {
  asm volatile("s_wait_asynccnt 0" ::: "memory");
}

#define WIN_SMEM 150176
#define GDFN_SMEM 107520

// ---------------------------------------------------------------------------
// Kernel 0: convert weights to f16 (with zero padding where needed)
// ---------------------------------------------------------------------------
__global__ __launch_bounds__(256) void k_cvt_weights(
    const float* __restrict__ qkvw, const float* __restrict__ projw,
    const float* __restrict__ pinw, const float* __restrict__ poutw,
    _Float16* __restrict__ qkvw16, _Float16* __restrict__ projw16,
    _Float16* __restrict__ pinw16, _Float16* __restrict__ poutw16) {
  int idx = blockIdx.x * 256 + threadIdx.x;
  if (idx < 288 * 96) qkvw16[idx] = (_Float16)qkvw[idx];
  if (idx < 96 * 96) projw16[idx] = (_Float16)projw[idx];
  if (idx < 512 * 96) {                      // pad rows 510..511 with zero
    int o = idx / 96, c = idx % 96;
    pinw16[idx] = (o < 510) ? (_Float16)pinw[o * 96 + c] : (_Float16)0.f;
  }
  if (idx < 96 * 256) {                      // pad K col 255 with zero
    int o = idx / 256, k = idx % 256;
    poutw16[idx] = (k < 255) ? (_Float16)poutw[o * 255 + k] : (_Float16)0.f;
  }
}

// ---------------------------------------------------------------------------
// Kernel 1: LayerNorm1 (per-pixel over C) + roll(-4,-4) + window partition
// out: xw[win=4096][token=64][c=96] f16
// ---------------------------------------------------------------------------
__global__ __launch_bounds__(256) void k_ln1_window(
    const float* __restrict__ x, const float* __restrict__ gam,
    const float* __restrict__ bet, _Float16* __restrict__ xw) {
  int p = blockIdx.x * 256 + threadIdx.x;    // 0..262143
  int bb = p >> 16, hw = p & 65535;
  int h = hw >> 8, w = hw & 255;
  const float* px = x + ((size_t)bb * 96) * 65536 + hw;
  float s = 0.f, s2 = 0.f;
  for (int c = 0; c < 96; ++c) {
    float v = px[(size_t)c * 65536];
    s += v; s2 += v * v;
  }
  float mu = s * (1.f / 96.f);
  float var = s2 * (1.f / 96.f) - mu * mu;
  float rs = rsqrtf(var + 1e-5f);
  int hs = (h + 252) & 255, wsi = (w + 252) & 255;   // shifted coords
  int wi = hs >> 3, ti = hs & 7, wj = wsi >> 3, tj = wsi & 7;
  size_t dst = (((size_t)((bb * 32 + wi) * 32 + wj)) * 64 + (ti * 8 + tj)) * 96;
  for (int c = 0; c < 96; ++c) {
    float v = (px[(size_t)c * 65536] - mu) * rs * gam[c] + bet[c];
    xw[dst + c] = (_Float16)v;
  }
}

// ---------------------------------------------------------------------------
// Kernel 2: fused window attention. 1 workgroup (8 waves) = 1 window.
// d_out = x + window_reverse(proj(softmax(qk^T+bias+mask) v))
// ---------------------------------------------------------------------------
__global__ __launch_bounds__(256) void k_win_attn(
    const float* __restrict__ xin, const _Float16* __restrict__ xw,
    const _Float16* __restrict__ qkvw, const _Float16* __restrict__ projw,
    const float* __restrict__ qkvb, const float* __restrict__ projb,
    const float* __restrict__ rpb, float* __restrict__ xout) {
  extern __shared__ char smem[];
  _Float16* sXW = (_Float16*)smem;                 // 64 x 104 (LDS offset 0)
  _Float16* sQK = sXW + 64 * 104;                  // 64 x 200 (q,k cols 0..191)
  _Float16* sVT = sQK + 64 * 200;                  // 96 x 72  (v transposed)
  float*    sS  = (float*)(sVT + 96 * 72);         // 3 x 64 x 68 scores
  _Float16* sP  = (_Float16*)(sS + 3 * 64 * 68);   // 3 x 64 x 72 softmax probs
  _Float16* sAO = sP + 3 * 64 * 72;                // 64 x 104 attention out
  float* sRPB = (float*)(sAO + 64 * 104);          // 675
  float* sQB  = sRPB + 676;                        // 288
  float* sPB  = sQB + 288;                         // 96

  int tid = threadIdx.x;
  int lane = tid & 31, wv = tid >> 5;
  int win = blockIdx.x;
  int bb = win >> 10;
  int wrem = win & 1023;
  int wi = wrem >> 5, wj = wrem & 31;

  // Warm the shared weight matrices into cache (L2-resident across all WGs).
  if (tid == 0) {
    __builtin_prefetch(qkvw, 0, 0);
    __builtin_prefetch(projw, 0, 0);
  }

  // ---- Phase A: async-stage window tokens (12KB) + small tables -------------
  // 64 rows x 192B from global (row stride 192B) -> LDS rows of stride 208B.
  {
    const char* src = (const char*)(xw + (size_t)win * 6144);
    for (int i = tid; i < 768; i += 256) {        // 768 x 16B chunks
      int n = i / 12, cp = i % 12;
      async_g2l_b128((unsigned)(n * 208 + cp * 16), src + i * 16);
    }
  }
  for (int i = tid; i < 675; i += 256) sRPB[i] = rpb[i];
  for (int i = tid; i < 288; i += 256) sQB[i] = qkvb[i];
  if (tid < 96) sPB[tid] = projb[tid];
  wait_async0();
  __syncthreads();

  // ---- Phase B: qkv = xw @ qkv_w^T + b  (M=64, N=288, K=96) -----------------
  for (int t = wv; t < 72; t += 8) {
    int mt = t & 3, nt = t >> 2;
    int m0 = mt * 16, n0 = nt * 16;
    v8f acc = {};
    for (int kk = 0; kk < 96; kk += 32) {
      v16h a = ld_frag(sXW, m0, 104, kk, lane);
      v16h b = ld_frag(qkvw, n0, 96, kk, lane);   // [o][c] = N-major
      acc = wmma_f16(a, b, acc);
    }
    int nn = n0 + (lane & 15);
    int mb = m0 + ((lane >> 4) << 3);
    float bias = sQB[nn];
    if (n0 < 192) {                 // q (0..95) and k (96..191) row-major
#pragma unroll
      for (int r = 0; r < 8; ++r)
        sQK[(mb + r) * 200 + nn] = (_Float16)(acc[r] + bias);
    } else {                        // v stored transposed: [h*32+d][token]
#pragma unroll
      for (int r = 0; r < 8; ++r)
        sVT[(nn - 192) * 72 + (mb + r)] = (_Float16)(acc[r] + bias);
    }
  }
  __syncthreads();

  // ---- Phase C: S = q @ k^T per head (K = 32 exactly one WMMA) --------------
  for (int t = wv; t < 48; t += 8) {
    int h = t / 16, r = t % 16;
    int mt = r & 3, nt = r >> 2;
    int m0 = mt * 16, n0 = nt * 16;
    v16h a = ld_frag(sQK, m0, 200, h * 32, lane);         // q
    v16h b = ld_frag(sQK, n0, 200, 96 + h * 32, lane);    // k ([tok][d] = N-major)
    v8f acc = {};
    acc = wmma_f16(a, b, acc);
    int nn = n0 + (lane & 15);
    int mb = m0 + ((lane >> 4) << 3);
#pragma unroll
    for (int r2 = 0; r2 < 8; ++r2) sS[(h * 64 + mb + r2) * 68 + nn] = acc[r2];
  }
  __syncthreads();

  // ---- Phase D: scale + rel-pos bias + shift mask + softmax -----------------
  if (tid < 192) {
    int h = tid / 64, n = tid % 64;
    int ti = n >> 3, tj = n & 7;
    int rn = regi(wi * 8 + ti) * 3 + regi(wj * 8 + tj);
    float* row = sS + (h * 64 + n) * 68;
    float mx = -3.0e38f;
    for (int m = 0; m < 64; ++m) {
      int si = m >> 3, sj = m & 7;
      float bias = sRPB[((ti - si + 7) * 15 + (tj - sj + 7)) * 3 + h];
      int rm = regi(wi * 8 + si) * 3 + regi(wj * 8 + sj);
      float v = row[m] * 0.17677669529663687f + bias +
                ((rn != rm) ? -100.f : 0.f);
      row[m] = v;
      mx = fmaxf(mx, v);
    }
    float sum = 0.f;
    _Float16* prow = sP + (h * 64 + n) * 72;
    for (int m = 0; m < 64; ++m) {
      float e = __expf(row[m] - mx);
      sum += e;
      prow[m] = (_Float16)e;
    }
    float inv = 1.f / sum;
    for (int m = 0; m < 64; ++m)
      prow[m] = (_Float16)((float)prow[m] * inv);
  }
  __syncthreads();

  // ---- Phase E: O = P @ v per head (M=64,N=32,K=64) -------------------------
  for (int t = wv; t < 24; t += 8) {
    int h = t / 8, r8 = t % 8;
    int mt = r8 >> 1, nt = r8 & 1;
    int m0 = mt * 16, n0 = nt * 16;
    v8f acc = {};
    for (int kk = 0; kk < 64; kk += 32) {
      v16h a = ld_frag(sP + h * 64 * 72, m0, 72, kk, lane);
      v16h b = ld_frag(sVT, h * 32 + n0, 72, kk, lane);   // [d][tok] = N-major
      acc = wmma_f16(a, b, acc);
    }
    int col = h * 32 + n0 + (lane & 15);
    int mb = m0 + ((lane >> 4) << 3);
#pragma unroll
    for (int r = 0; r < 8; ++r) sAO[(mb + r) * 104 + col] = (_Float16)acc[r];
  }
  __syncthreads();

  // ---- Phase F: proj + window reverse + roll(+4) + residual -----------------
  for (int t = wv; t < 24; t += 8) {
    int mt = t & 3, nt = t >> 2;
    int m0 = mt * 16, n0 = nt * 16;
    v8f acc = {};
    for (int kk = 0; kk < 96; kk += 32) {
      v16h a = ld_frag(sAO, m0, 104, kk, lane);
      v16h b = ld_frag(projw, n0, 96, kk, lane);          // [o][i] = N-major
      acc = wmma_f16(a, b, acc);
    }
    int o = n0 + (lane & 15);
    int mb = m0 + ((lane >> 4) << 3);
    float pb = sPB[o];
#pragma unroll
    for (int r = 0; r < 8; ++r) {
      int n = mb + r;
      int hh = (wi * 8 + (n >> 3) + 4) & 255;
      int ww = (wj * 8 + (n & 7) + 4) & 255;
      size_t idx = (((size_t)bb * 96 + o) * 256 + hh) * 256 + ww;
      xout[idx] = xin[idx] + acc[r] + pb;
    }
  }
}

// ---------------------------------------------------------------------------
// Kernel 3: LayerNorm2 over C, stored flat in [B,HW,C] order (f16).
// Consumer re-reads the same buffer flat as [B,C,H,W] — the faithful raw
// reshape of the reference falls out of the index reinterpretation.
// ---------------------------------------------------------------------------
__global__ __launch_bounds__(256) void k_ln2_raw(
    const float* __restrict__ x1, const float* __restrict__ gam,
    const float* __restrict__ bet, _Float16* __restrict__ yln) {
  int p = blockIdx.x * 256 + threadIdx.x;
  int bb = p >> 16, hw = p & 65535;
  const float* px = x1 + ((size_t)bb * 96) * 65536 + hw;
  float s = 0.f, s2 = 0.f;
  for (int c = 0; c < 96; ++c) {
    float v = px[(size_t)c * 65536];
    s += v; s2 += v * v;
  }
  float mu = s * (1.f / 96.f);
  float var = s2 * (1.f / 96.f) - mu * mu;
  float rs = rsqrtf(var + 1e-5f);
  size_t dst = ((size_t)bb * 65536 + hw) * 96;
  for (int c = 0; c < 96; ++c) {
    float v = (px[(size_t)c * 65536] - mu) * rs * gam[c] + bet[c];
    yln[dst + c] = (_Float16)v;
  }
}

// ---------------------------------------------------------------------------
// Kernel 5: fused GDFN. 1 workgroup = 8x8 pixel tile (10x10 halo).
// pin 1x1 (96->510, recomputed over halo) -> dw3x3 + GELU gate -> pout 1x1
// (255->96), accumulated into d_out which already holds x1.
// ---------------------------------------------------------------------------
__global__ __launch_bounds__(256) void k_gdfn(
    const _Float16* __restrict__ yln, const _Float16* __restrict__ pinw,
    const float* __restrict__ dww, const _Float16* __restrict__ poutw,
    float* __restrict__ xio) {
  extern __shared__ char smem[];
  _Float16* sY  = (_Float16*)smem;            // 112 x 96  (halo pixels x c)
  _Float16* sY2 = sY + 112 * 96;              // 256 x 104 (chunk channels x halo px)
  _Float16* sG  = sY2 + 256 * 104;            // 64 x 256  (pixel x gate channel)

  int tid = threadIdx.x;
  int lane = tid & 31, wv = tid >> 5;
  int b = blockIdx.x >> 10;
  int tno = blockIdx.x & 1023;
  int r0 = (tno >> 5) * 8, c0 = (tno & 31) * 8;

  if (tid == 0) {
    __builtin_prefetch(pinw, 0, 0);
    __builtin_prefetch(poutw, 0, 0);
  }

  // stage halo tile of (raw-reshaped) LN2 output: sY[p_halo][c]
  for (int i = tid; i < 9600; i += 256) {
    int c = i / 100, hp = i % 100;
    int hy = r0 - 1 + hp / 10;
    int hx = c0 - 1 + hp % 10;
    _Float16 v = (_Float16)0.f;
    if (hy >= 0 && hy < 256 && hx >= 0 && hx < 256)
      v = yln[(size_t)b * 6291456 + (size_t)c * 65536 + hy * 256 + hx];
    sY[hp * 96 + c] = v;
  }
  for (int i = tid; i < 12 * 96; i += 256)    // N-pad rows 100..111
    sY[(100 + i / 96) * 96 + (i % 96)] = (_Float16)0.f;
  if (tid < 64) sG[tid * 256 + 255] = (_Float16)0.f;  // K-pad for pout
  __syncthreads();

  // two chunks of 128 gate channels to keep y2 resident in LDS
  for (int chunk = 0; chunk < 2; ++chunk) {
    // pin GEMM: rows = pin channels (low half + high half), cols = halo px
    for (int t = wv; t < 112; t += 8) {
      int mt = t % 16, nt = t / 16;
      int m0g = (mt < 8) ? (chunk * 128 + mt * 16)
                         : (255 + chunk * 128 + (mt - 8) * 16);
      int m0l = mt * 16;
      int n0 = nt * 16;
      v8f acc = {};
      for (int kk = 0; kk < 96; kk += 32) {
        v16h a = ld_frag(pinw, m0g, 96, kk, lane);   // [o][c]
        v16h bf = ld_frag(sY, n0, 96, kk, lane);     // [p][c] = N-major
        acc = wmma_f16(a, bf, acc);
      }
      int nn = n0 + (lane & 15);
      int mb = m0l + ((lane >> 4) << 3);
      if (nn < 104) {
#pragma unroll
        for (int r = 0; r < 8; ++r)
          sY2[(mb + r) * 104 + nn] = (_Float16)acc[r];
      }
    }
    __syncthreads();

    // depthwise 3x3 + exact GELU gate
    int base_o = chunk * 128;
    int cnt = chunk ? 127 : 128;
    for (int i = tid; i < cnt * 64; i += 256) {
      int lo = i >> 6, p = i & 63;
      int oo = base_o + lo;
      int hr = p >> 3, hc = p & 7;
      float a1 = 0.f, a2 = 0.f;
#pragma unroll
      for (int dy = 0; dy < 3; ++dy)
#pragma unroll
        for (int dx = 0; dx < 3; ++dx) {
          int hp = (hr + dy) * 10 + (hc + dx);
          a1 += dww[oo * 9 + dy * 3 + dx] * (float)sY2[lo * 104 + hp];
          a2 += dww[(oo + 255) * 9 + dy * 3 + dx] *
                (float)sY2[(128 + lo) * 104 + hp];
        }
      float ge = 0.5f * a1 * (1.f + erff(a1 * 0.70710678118654752f));
      sG[p * 256 + oo] = (_Float16)(ge * a2);
    }
    __syncthreads();
  }

  // pout GEMM: M=96 channels, N=64 pixels, K=256 (padded) -> accumulate
  for (int t = wv; t < 24; t += 8) {
    int mt = t % 6, nt = t / 6;
    int m0 = mt * 16, n0 = nt * 16;
    v8f acc = {};
    for (int kk = 0; kk < 256; kk += 32) {
      v16h a = ld_frag(poutw, m0, 256, kk, lane);    // [o][k]
      v16h bf = ld_frag(sG, n0, 256, kk, lane);      // [p][k] = N-major
      acc = wmma_f16(a, bf, acc);
    }
    int p = n0 + (lane & 15);
    int hh = r0 + (p >> 3), ww = c0 + (p & 7);
    int mb = m0 + ((lane >> 4) << 3);
#pragma unroll
    for (int r = 0; r < 8; ++r) {
      size_t idx = (((size_t)b * 96 + (mb + r)) * 256 + hh) * 256 + ww;
      xio[idx] += acc[r];
    }
  }
}

// ---------------------------------------------------------------------------
extern "C" void kernel_launch(void* const* d_in, const int* in_sizes, int n_in,
                              void* d_out, int out_size, void* d_ws,
                              size_t ws_size, hipStream_t stream) {
  const float* x     = (const float*)d_in[0];
  const float* rpb   = (const float*)d_in[1];
  const float* qkvw  = (const float*)d_in[2];
  const float* qkvb  = (const float*)d_in[3];
  const float* projw = (const float*)d_in[4];
  const float* projb = (const float*)d_in[5];
  const float* n1g   = (const float*)d_in[6];
  const float* n1b   = (const float*)d_in[7];
  const float* n2g   = (const float*)d_in[8];
  const float* n2b   = (const float*)d_in[9];
  const float* pinw  = (const float*)d_in[10];
  const float* dww   = (const float*)d_in[11];
  const float* poutw = (const float*)d_in[12];
  float* out = (float*)d_out;

  char* ws = (char*)d_ws;
  _Float16* xw = (_Float16*)ws;        ws += 50331648;   // 4096*64*96 f16
  _Float16* yln = (_Float16*)ws;       ws += 50331648;   // 4*65536*96 f16
  _Float16* qkvw16 = (_Float16*)ws;    ws += 55296;      // 288*96
  _Float16* projw16 = (_Float16*)ws;   ws += 18688;      // 96*96 (padded)
  _Float16* pinw16 = (_Float16*)ws;    ws += 98304;      // 512*96
  _Float16* poutw16 = (_Float16*)ws;   ws += 49152;      // 96*256

  k_cvt_weights<<<192, 256, 0, stream>>>(qkvw, projw, pinw, poutw,
                                         qkvw16, projw16, pinw16, poutw16);
  k_ln1_window<<<1024, 256, 0, stream>>>(x, n1g, n1b, xw);
  k_win_attn<<<4096, 256, WIN_SMEM, stream>>>(x, xw, qkvw16, projw16,
                                              qkvb, projb, rpb, out);
  k_ln2_raw<<<1024, 256, 0, stream>>>(out, n2g, n2b, yln);
  k_gdfn<<<4096, 256, GDFN_SMEM, stream>>>(yln, pinw16, dww, poutw16, out);
}
